// GrassmanianformerClassifier_44753559225155
// MI455X (gfx1250) — compile-verified, matrix operand order checked by reading, not run on hardware
//
#include <hip/hip_runtime.h>
#include <hip/hip_bf16.h>
#include <math.h>

// ---------------- problem constants ----------------
constexpr int B0 = 32, N0 = 196, D0 = 768, H0 = 12, HD0 = 64, L0 = 12,
              FF0 = 3072, CLS0 = 1000;
constexpr int BND = B0 * N0 * D0;          // 4,816,896
constexpr int BN3D = B0 * N0 * 3 * D0;     // 14,450,688
constexpr int BNFF = B0 * N0 * FF0;        // 19,267,584
constexpr int HNN = H0 * N0 * N0;          // 460,992
constexpr int KP = 224;                    // N0 padded to 32 for attn@v K-dim
constexpr float EPS = 1e-5f;

// ---------------- WMMA types / helpers ----------------
typedef __attribute__((ext_vector_type(16))) __bf16 bf16x16;
typedef __attribute__((ext_vector_type(8)))  float  f32x8;

__device__ __forceinline__ f32x8 wmma_bf16(bf16x16 a, bf16x16 b, f32x8 c) {
  return __builtin_amdgcn_wmma_f32_16x16x32_bf16(
      false, a, false, b, (short)0, c, false, false);
}

// ---------------- async global->LDS copy (CDNA5 ASYNCcnt path) ----------------
typedef int v4i __attribute__((vector_size(16)));
typedef __attribute__((address_space(3))) v4i v4i_lds;

#if defined(__has_builtin)
#if __has_builtin(__builtin_amdgcn_global_load_async_to_lds_b128)
#define ASYNC_LDS 1
#endif
#endif

__device__ __forceinline__ void async_cp_b128(__bf16* dst, const __bf16* src) {
#ifdef ASYNC_LDS
  __builtin_amdgcn_global_load_async_to_lds_b128((v4i*)src, (v4i_lds*)dst, 0,
                                                 0);
#else
  *(uint4*)dst = *(const uint4*)src;  // sync fallback
#endif
}

__device__ __forceinline__ void wait_async0() {
#ifdef ASYNC_LDS
#if __has_builtin(__builtin_amdgcn_s_wait_asynccnt)
  __builtin_amdgcn_s_wait_asynccnt(0);
#else
  asm volatile("s_wait_asynccnt 0x0" ::: "memory");
#endif
#endif
}

// CDNA5 16-bit matrix fragment: lane = r + 16*g,
//  g==0 : K = {0..7, 16..23},  g==1 : K = {8..15, 24..31}
__device__ __forceinline__ bf16x16 load_frag(const __bf16* lds, int row0,
                                             int ld, int k0) {
  const int lane = threadIdx.x & 31;
  const __bf16* p = lds + (row0 + (lane & 15)) * ld + k0 + (lane >> 4) * 8;
  bf16x16 f;
#pragma unroll
  for (int i = 0; i < 8; ++i) f[i] = p[i];
#pragma unroll
  for (int i = 0; i < 8; ++i) f[8 + i] = p[16 + i];
  return f;
}

// ---------------- generic WMMA GEMM ----------------
// A[M,K] bf16 row-major; Bt[N,K] bf16 (K-contiguous); C = epi(A@B + bias)(+res)
// act: 0 none, 1 exact GELU. out_bf16: C dtype. Double-buffered async pipeline.
#define GM_BM 128
#define GM_BN 64
#define GM_BK 64
#define GM_LD (GM_BK + 8)
__global__ __launch_bounds__(256) void k_gemm(
    const __bf16* __restrict__ A, const __bf16* __restrict__ Bt,
    const float* __restrict__ bias, const float* __restrict__ res,
    void* __restrict__ Cv, int M, int N, int K, int act, int out_bf16) {
  __shared__ __bf16 As[2 * GM_BM * GM_LD];
  __shared__ __bf16 Bs[2 * GM_BN * GM_LD];
  const int tid = threadIdx.x;
  const int wave = tid >> 5;
  const int wm = wave >> 1;  // 0..3 -> 32 rows
  const int wn = wave & 1;   // 0..1 -> 32 cols
  const int m0 = blockIdx.y * GM_BM;
  const int n0 = blockIdx.x * GM_BN;
  const bool interior = (m0 + GM_BM <= M) && (n0 + GM_BN <= N);

  f32x8 acc[2][2];
  f32x8 zero = {0.f, 0.f, 0.f, 0.f, 0.f, 0.f, 0.f, 0.f};
#pragma unroll
  for (int i = 0; i < 2; ++i)
#pragma unroll
    for (int j = 0; j < 2; ++j) acc[i][j] = zero;

  auto issue = [&](int k0, int bufi) {
    __bf16* Ab = As + bufi * (GM_BM * GM_LD);
    __bf16* Bb = Bs + bufi * (GM_BN * GM_LD);
#pragma unroll
    for (int i = 0; i < 4; ++i) {  // A: 128 rows x 8 b128-chunks
      int q = tid + 256 * i;
      int r = q >> 3, c = (q & 7) * 8;
      async_cp_b128(Ab + r * GM_LD + c, A + (size_t)(m0 + r) * K + k0 + c);
    }
#pragma unroll
    for (int i = 0; i < 2; ++i) {  // B: 64 rows x 8 chunks
      int q = tid + 256 * i;
      int r = q >> 3, c = (q & 7) * 8;
      async_cp_b128(Bb + r * GM_LD + c, Bt + (size_t)(n0 + r) * K + k0 + c);
    }
  };
  auto compute = [&](int bufi) {
    const __bf16* Ab = As + bufi * (GM_BM * GM_LD);
    const __bf16* Bb = Bs + bufi * (GM_BN * GM_LD);
#pragma unroll
    for (int ks = 0; ks < GM_BK; ks += 32) {
      bf16x16 a0 = load_frag(Ab, wm * 32, GM_LD, ks);
      bf16x16 a1 = load_frag(Ab, wm * 32 + 16, GM_LD, ks);
      bf16x16 b0 = load_frag(Bb, wn * 32, GM_LD, ks);
      bf16x16 b1 = load_frag(Bb, wn * 32 + 16, GM_LD, ks);
      acc[0][0] = wmma_bf16(a0, b0, acc[0][0]);
      acc[0][1] = wmma_bf16(a0, b1, acc[0][1]);
      acc[1][0] = wmma_bf16(a1, b0, acc[1][0]);
      acc[1][1] = wmma_bf16(a1, b1, acc[1][1]);
    }
  };

  if (interior && (K % GM_BK == 0)) {
    const int S = K / GM_BK;
    issue(0, 0);
    wait_async0();
    __syncthreads();
    for (int s = 0; s < S; ++s) {
      const int cur = s & 1;
      if (s + 1 < S) issue((s + 1) * GM_BK, cur ^ 1);
      compute(cur);
      if (s + 1 < S) {
        wait_async0();
        __syncthreads();
      }
    }
  } else {
    for (int k0 = 0; k0 < K; k0 += GM_BK) {
      for (int i = 0; i < 32; ++i) {
        int q = tid + 256 * i;
        int r = q >> 6, c = q & 63;
        int gm = m0 + r, gk = k0 + c;
        As[r * GM_LD + c] =
            (gm < M && gk < K) ? A[(size_t)gm * K + gk] : (__bf16)0.f;
      }
      for (int i = 0; i < 16; ++i) {
        int q = tid + 256 * i;
        int r = q >> 6, c = q & 63;
        int gn = n0 + r, gk = k0 + c;
        Bs[r * GM_LD + c] =
            (gn < N && gk < K) ? Bt[(size_t)gn * K + gk] : (__bf16)0.f;
      }
      __syncthreads();
      compute(0);
      __syncthreads();
    }
  }

  // C layout: VGPR r -> M = base + r + 8*g, N = lane&15
  const int lane = tid & 31;
  const int cn = lane & 15;
  const int g = lane >> 4;
#pragma unroll
  for (int ti = 0; ti < 2; ++ti)
#pragma unroll
    for (int tj = 0; tj < 2; ++tj) {
      int rb = m0 + wm * 32 + ti * 16 + 8 * g;
      int cb = n0 + wn * 32 + tj * 16 + cn;
      if (cb < N) {
        float bv = bias ? bias[cb] : 0.f;
#pragma unroll
        for (int r = 0; r < 8; ++r) {
          int m = rb + r;
          if (m < M) {
            float v = acc[ti][tj][r] + bv;
            if (act == 1) v = 0.5f * v * (1.f + erff(v * 0.70710678118f));
            if (res) v += res[(size_t)m * N + cb];
            if (out_bf16)
              ((__bf16*)Cv)[(size_t)m * N + cb] = (__bf16)v;
            else
              ((float*)Cv)[(size_t)m * N + cb] = v;
          }
        }
      }
    }
}

// ---------------- strided-batched WMMA GEMM ----------------
// A[M,K] bf16 (row stride = K); Bt[N,K] bf16; C[M,N].
#define BG_BM 64
#define BG_BN 64
#define BG_BK 32
#define BG_LD (BG_BK + 8)
__global__ __launch_bounds__(256) void k_gemm_batched(
    const __bf16* __restrict__ A, const __bf16* __restrict__ Bt,
    void* __restrict__ Cv, int M, int N, int K, long strideA, long strideB,
    long strideC, int out_bf16) {
  const int z = blockIdx.z;
  A += (size_t)z * strideA;
  Bt += (size_t)z * strideB;
  __shared__ __bf16 As[2 * BG_BM * BG_LD];
  __shared__ __bf16 Bs[2 * BG_BN * BG_LD];
  const int tid = threadIdx.x;
  const int wave = tid >> 5;
  const int wm = wave >> 1;  // 0..3 -> 16 rows
  const int wn = wave & 1;   // 0..1 -> 32 cols
  const int m0 = blockIdx.y * BG_BM;
  const int n0 = blockIdx.x * BG_BN;
  const bool interior = (m0 + BG_BM <= M) && (n0 + BG_BN <= N);

  f32x8 acc0 = {0.f, 0.f, 0.f, 0.f, 0.f, 0.f, 0.f, 0.f};
  f32x8 acc1 = acc0;

  auto issue = [&](int k0, int bufi) {
    __bf16* Ab = As + bufi * (BG_BM * BG_LD);
    __bf16* Bb = Bs + bufi * (BG_BN * BG_LD);
    {  // A: 64 rows x 4 chunks = 256
      int r = tid >> 2, c = (tid & 3) * 8;
      async_cp_b128(Ab + r * BG_LD + c, A + (size_t)(m0 + r) * K + k0 + c);
    }
    {  // B: 64 rows x 4 chunks
      int r = tid >> 2, c = (tid & 3) * 8;
      async_cp_b128(Bb + r * BG_LD + c, Bt + (size_t)(n0 + r) * K + k0 + c);
    }
  };
  auto compute = [&](int bufi) {
    const __bf16* Ab = As + bufi * (BG_BM * BG_LD);
    const __bf16* Bb = Bs + bufi * (BG_BN * BG_LD);
    bf16x16 a = load_frag(Ab, wm * 16, BG_LD, 0);
    bf16x16 b0 = load_frag(Bb, wn * 32, BG_LD, 0);
    bf16x16 b1 = load_frag(Bb, wn * 32 + 16, BG_LD, 0);
    acc0 = wmma_bf16(a, b0, acc0);
    acc1 = wmma_bf16(a, b1, acc1);
  };

  if (interior) {  // K is always a multiple of 32 here (64 or 224)
    const int S = K / BG_BK;
    issue(0, 0);
    wait_async0();
    __syncthreads();
    for (int s = 0; s < S; ++s) {
      const int cur = s & 1;
      if (s + 1 < S) issue((s + 1) * BG_BK, cur ^ 1);
      compute(cur);
      if (s + 1 < S) {
        wait_async0();
        __syncthreads();
      }
    }
  } else {
    for (int k0 = 0; k0 < K; k0 += BG_BK) {
      for (int i = 0; i < 8; ++i) {
        int q = tid + 256 * i;
        int r = q >> 5, c = q & 31;
        int gm = m0 + r, gk = k0 + c;
        As[r * BG_LD + c] =
            (gm < M && gk < K) ? A[(size_t)gm * K + gk] : (__bf16)0.f;
      }
      for (int i = 0; i < 8; ++i) {
        int q = tid + 256 * i;
        int r = q >> 5, c = q & 31;
        int gn = n0 + r, gk = k0 + c;
        Bs[r * BG_LD + c] =
            (gn < N && gk < K) ? Bt[(size_t)gn * K + gk] : (__bf16)0.f;
      }
      __syncthreads();
      compute(0);
      __syncthreads();
    }
  }

  const int lane = tid & 31;
  const int cn = lane & 15;
  const int g = lane >> 4;
  int rb = m0 + wm * 16 + 8 * g;
#pragma unroll
  for (int tj = 0; tj < 2; ++tj) {
    int cb = n0 + wn * 32 + tj * 16 + cn;
    const f32x8& a = tj ? acc1 : acc0;
    if (cb < N) {
#pragma unroll
      for (int r = 0; r < 8; ++r) {
        int m = rb + r;
        if (m < M) {
          size_t off = (size_t)z * strideC + (size_t)m * N + cb;
          if (out_bf16)
            ((__bf16*)Cv)[off] = (__bf16)a[r];
          else
            ((float*)Cv)[off] = a[r];
        }
      }
    }
  }
}

// ---------------- weight convert + transpose: W[K,N] f32 -> WT[N,K] bf16 ----
__global__ __launch_bounds__(256) void k_cvt_transpose(
    const float* __restrict__ W, __bf16* __restrict__ WT, int K, int N) {
  __shared__ float tile[32][33];
  const int z = blockIdx.z;
  W += (size_t)z * K * N;
  WT += (size_t)z * K * N;
  const int n0 = blockIdx.x * 32, k0 = blockIdx.y * 32;
  const int tx = threadIdx.x & 31, ty = threadIdx.x >> 5;
#pragma unroll
  for (int j = 0; j < 4; ++j) {
    int k = k0 + ty + 8 * j, n = n0 + tx;
    tile[ty + 8 * j][tx] = (k < K && n < N) ? W[(size_t)k * N + n] : 0.f;
  }
  __syncthreads();
#pragma unroll
  for (int j = 0; j < 4; ++j) {
    int n = n0 + ty + 8 * j, k = k0 + tx;
    if (n < N && k < K) WT[(size_t)n * K + k] = (__bf16)tile[tx][ty + 8 * j];
  }
}

// ---------------- elementwise / reduction kernels ----------------
__global__ void k_add_pos(const float* __restrict__ x,
                          const float* __restrict__ pos,
                          float* __restrict__ out) {
  size_t i = (size_t)blockIdx.x * blockDim.x + threadIdx.x;
  if (i >= (size_t)BND) return;
  size_t nd = i % ((size_t)N0 * D0);
  out[i] = x[i] + pos[nd];
}

// LN over rows; optional residual; optional fp32 and bf16 outputs.
__global__ __launch_bounds__(256) void k_layernorm(
    const float* __restrict__ in, const float* __restrict__ res,
    const float* __restrict__ w, const float* __restrict__ b,
    float* __restrict__ outf, __bf16* __restrict__ outb, int cols) {
  const int row = blockIdx.x;
  const float* ip = in + (size_t)row * cols;
  const float* rp = res ? res + (size_t)row * cols : nullptr;
  __shared__ float red[256];
  float s = 0.f, ss = 0.f;
  for (int c = threadIdx.x; c < cols; c += 256) {
    float v = ip[c] + (rp ? rp[c] : 0.f);
    s += v;
    ss += v * v;
  }
  red[threadIdx.x] = s;
  __syncthreads();
  for (int o = 128; o > 0; o >>= 1) {
    if (threadIdx.x < o) red[threadIdx.x] += red[threadIdx.x + o];
    __syncthreads();
  }
  float mean = red[0] / cols;
  __syncthreads();
  red[threadIdx.x] = ss;
  __syncthreads();
  for (int o = 128; o > 0; o >>= 1) {
    if (threadIdx.x < o) red[threadIdx.x] += red[threadIdx.x + o];
    __syncthreads();
  }
  float var = red[0] / cols - mean * mean;
  float inv = rsqrtf(var + EPS);
  for (int c = threadIdx.x; c < cols; c += 256) {
    float v = ip[c] + (rp ? rp[c] : 0.f);
    float y = (v - mean) * inv * w[c] + b[c];
    if (outf) outf[(size_t)row * cols + c] = y;
    if (outb) outb[(size_t)row * cols + c] = (__bf16)y;
  }
}

// qkv (B,N,3,H,HD) f32 -> qq/kk [b,h,n,d] bf16 (row=K-contig), vvt [b,h,d,n(KP)]
__global__ __launch_bounds__(64) void k_qk_prep(const float* __restrict__ qkv,
                                                __bf16* __restrict__ qq,
                                                __bf16* __restrict__ kk,
                                                __bf16* __restrict__ vvt) {
  const int bid = blockIdx.x;
  const int b = bid / (H0 * N0);
  const int r = bid % (H0 * N0);
  const int h = r / N0;
  const int n = r % N0;
  const int t = threadIdx.x;  // 0..63
  __shared__ float red[64];
  size_t base = ((size_t)b * N0 + n) * (3 * D0) + h * HD0;
  float qv = qkv[base + 0 * D0 + t];
  float kv = qkv[base + 1 * D0 + t];
  float vva = qkv[base + 2 * D0 + t];
  red[t] = qv * qv;
  __syncthreads();
  for (int o = 32; o > 0; o >>= 1) {
    if (t < o) red[t] += red[t + o];
    __syncthreads();
  }
  float qs = red[0];
  __syncthreads();
  red[t] = kv * kv;
  __syncthreads();
  for (int o = 32; o > 0; o >>= 1) {
    if (t < o) red[t] += red[t + o];
    __syncthreads();
  }
  float ks = red[0];
  const int bh = b * H0 + h;
  size_t ob = ((size_t)bh * N0 + n) * HD0 + t;
  qq[ob] = (__bf16)(qv * qv / qs);
  kk[ob] = (__bf16)(kv * kv / ks);
  vvt[((size_t)bh * HD0 + t) * KP + n] = (__bf16)vva;
}

__global__ void k_zero(float* p, int n) {
  int i = blockIdx.x * blockDim.x + threadIdx.x;
  if (i < n) p[i] = 0.f;
}
__global__ void k_zero_bf16(__bf16* p, size_t n) {
  size_t i = (size_t)blockIdx.x * blockDim.x + threadIdx.x;
  if (i < n) p[i] = (__bf16)0.f;
}

__global__ __launch_bounds__(256) void k_attn_stats(
    const float* __restrict__ attn, float* __restrict__ stats) {
  const int b = blockIdx.y;
  size_t base = (size_t)b * HNN;
  float s = 0.f, ss = 0.f;
  for (int i = blockIdx.x * 256 + threadIdx.x; i < HNN; i += gridDim.x * 256) {
    float v = attn[base + i];
    s += v;
    ss += v * v;
  }
  __shared__ float r1[256], r2[256];
  r1[threadIdx.x] = s;
  r2[threadIdx.x] = ss;
  __syncthreads();
  for (int o = 128; o > 0; o >>= 1) {
    if (threadIdx.x < o) {
      r1[threadIdx.x] += r1[threadIdx.x + o];
      r2[threadIdx.x] += r2[threadIdx.x + o];
    }
    __syncthreads();
  }
  if (threadIdx.x == 0) {
    atomicAdd(&stats[2 * b], r1[0]);
    atomicAdd(&stats[2 * b + 1], r2[0]);
  }
}

// normalize attn in-place (fp32, d_out) + emit bf16 copy with K padded to KP
__global__ void k_attn_norm(float* __restrict__ attn,
                            const float* __restrict__ stats,
                            const float* __restrict__ aw,
                            const float* __restrict__ ab,
                            __bf16* __restrict__ attn_bf) {
  size_t idx = (size_t)blockIdx.x * blockDim.x + threadIdx.x;
  size_t total = (size_t)B0 * HNN;
  if (idx >= total) return;
  int b = (int)(idx / HNN);
  int r = (int)(idx % HNN);
  float mu = stats[2 * b] / (float)HNN;
  float var = stats[2 * b + 1] / (float)HNN - mu * mu;
  float inv = rsqrtf(var + EPS);
  float v = (attn[idx] - mu) * inv * aw[r] + ab[r];
  attn[idx] = v;
  int row = r / N0, j = r % N0;  // row = h*N0 + i
  attn_bf[((size_t)b * (H0 * N0) + row) * KP + j] = (__bf16)v;
}

// (B,H,N,HD) bf16 -> (B,N,D) bf16
__global__ void k_merge_heads(const __bf16* __restrict__ oh,
                              __bf16* __restrict__ out) {
  size_t i = (size_t)blockIdx.x * blockDim.x + threadIdx.x;
  if (i >= (size_t)BND) return;
  int d = (int)(i % D0);
  int n = (int)((i / D0) % N0);
  int b = (int)(i / ((size_t)N0 * D0));
  int h = d / HD0, dd = d % HD0;
  out[i] = oh[(((size_t)b * H0 + h) * N0 + n) * HD0 + dd];
}

__global__ __launch_bounds__(256) void k_pool_score(
    const float* __restrict__ xf, const float* __restrict__ pw,
    const float* __restrict__ pb, float* __restrict__ s) {
  const int row = blockIdx.x;
  __shared__ float red[256];
  float acc = 0.f;
  for (int c = threadIdx.x; c < D0; c += 256)
    acc += xf[(size_t)row * D0 + c] * pw[c];
  red[threadIdx.x] = acc;
  __syncthreads();
  for (int o = 128; o > 0; o >>= 1) {
    if (threadIdx.x < o) red[threadIdx.x] += red[threadIdx.x + o];
    __syncthreads();
  }
  if (threadIdx.x == 0) s[row] = red[0] + pb[0];
}

__global__ __launch_bounds__(256) void k_softmax_pool(
    const float* __restrict__ xf, const float* __restrict__ s,
    __bf16* __restrict__ pooled) {
  const int b = blockIdx.x;
  __shared__ float a[N0];
  __shared__ float red[256];
  const int t = threadIdx.x;
  float m = -1e30f;
  for (int n = t; n < N0; n += 256) m = fmaxf(m, s[b * N0 + n]);
  red[t] = m;
  __syncthreads();
  for (int o = 128; o > 0; o >>= 1) {
    if (t < o) red[t] = fmaxf(red[t], red[t + o]);
    __syncthreads();
  }
  float mx = red[0];
  __syncthreads();
  float sum = 0.f;
  for (int n = t; n < N0; n += 256) {
    float e = expf(s[b * N0 + n] - mx);
    a[n] = e;
    sum += e;
  }
  red[t] = sum;
  __syncthreads();
  for (int o = 128; o > 0; o >>= 1) {
    if (t < o) red[t] += red[t + o];
    __syncthreads();
  }
  float tot = red[0];
  __syncthreads();
  for (int n = t; n < N0; n += 256) a[n] /= tot;
  __syncthreads();
  for (int d = t; d < D0; d += 256) {
    float acc = 0.f;
    for (int n = 0; n < N0; ++n) acc += a[n] * xf[((size_t)b * N0 + n) * D0 + d];
    pooled[(size_t)b * D0 + d] = (__bf16)acc;
  }
}

// ---------------- host orchestration ----------------
static inline dim3 g1(size_t n, int blk) { return dim3((unsigned)((n + blk - 1) / blk)); }

extern "C" void kernel_launch(void* const* d_in, const int* in_sizes, int n_in,
                              void* d_out, int out_size, void* d_ws,
                              size_t ws_size, hipStream_t stream) {
  (void)in_sizes; (void)n_in; (void)out_size; (void)ws_size;
  const float* x      = (const float*)d_in[0];
  const float* pos    = (const float*)d_in[1];
  const float* qkv_w  = (const float*)d_in[2];
  const float* qkv_b  = (const float*)d_in[3];
  const float* proj_w = (const float*)d_in[4];
  const float* proj_b = (const float*)d_in[5];
  const float* lin1_w = (const float*)d_in[6];
  const float* lin1_b = (const float*)d_in[7];
  const float* lin2_w = (const float*)d_in[8];
  const float* lin2_b = (const float*)d_in[9];
  const float* pre_w  = (const float*)d_in[10];
  const float* pre_b  = (const float*)d_in[11];
  const float* n1_w   = (const float*)d_in[12];
  const float* n1_b   = (const float*)d_in[13];
  const float* aln_w  = (const float*)d_in[14];
  const float* aln_b  = (const float*)d_in[15];
  const float* fin_w  = (const float*)d_in[16];
  const float* fin_b  = (const float*)d_in[17];
  const float* pool_w = (const float*)d_in[18];
  const float* pool_b = (const float*)d_in[19];
  const float* fc_w   = (const float*)d_in[20];
  const float* fc_b   = (const float*)d_in[21];

  float* logits = (float*)d_out;
  float* attn   = logits + (size_t)B0 * CLS0;  // last-layer attn lives in d_out

  char* cur = (char*)d_ws;
  auto alloc = [&](size_t bytes) {
    char* p = cur;
    cur += (bytes + 255) & ~(size_t)255;
    return p;
  };
  // fp32 buffers
  float*  srcA    = (float*)alloc((size_t)BND * 4);
  float*  srcB    = (float*)alloc((size_t)BND * 4);
  float*  bufQKV  = (float*)alloc((size_t)BN3D * 4);  // reused: proj out, xf
  float*  stats   = (float*)alloc(64 * 4);
  float*  bufS    = (float*)alloc((size_t)B0 * N0 * 4);
  // bf16 buffers
  __bf16* bufY    = (__bf16*)alloc((size_t)BND * 2);  // LN out / oh reuse
  __bf16* srcBb   = (__bf16*)alloc((size_t)BND * 2);
  __bf16* qq      = (__bf16*)alloc((size_t)BND * 2);  // reused: merged heads
  __bf16* kk      = (__bf16*)alloc((size_t)BND * 2);
  __bf16* vvt     = (__bf16*)alloc((size_t)B0 * H0 * HD0 * KP * 2);
  __bf16* attnbf  = (__bf16*)alloc((size_t)B0 * H0 * N0 * KP * 2);
  __bf16* bufFF   = (__bf16*)alloc((size_t)BNFF * 2);
  __bf16* pooled  = (__bf16*)alloc((size_t)B0 * D0 * 2);
  // bf16 transposed weights [N,K]
  __bf16* qkv_wt  = (__bf16*)alloc((size_t)L0 * 3 * D0 * D0 * 2);
  __bf16* proj_wt = (__bf16*)alloc((size_t)L0 * D0 * D0 * 2);
  __bf16* lin1_wt = (__bf16*)alloc((size_t)L0 * FF0 * D0 * 2);
  __bf16* lin2_wt = (__bf16*)alloc((size_t)L0 * D0 * FF0 * 2);
  __bf16* fc_wt   = (__bf16*)alloc((size_t)CLS0 * D0 * 2);

  const int ROWS = B0 * N0;  // 6272
  const int BH = B0 * H0;    // 384

  // ---- one-time per launch: weight convert+transpose, zero padded bufs ----
  k_cvt_transpose<<<dim3(3 * D0 / 32, D0 / 32, L0), 256, 0, stream>>>(
      qkv_w, qkv_wt, D0, 3 * D0);
  k_cvt_transpose<<<dim3(D0 / 32, D0 / 32, L0), 256, 0, stream>>>(
      proj_w, proj_wt, D0, D0);
  k_cvt_transpose<<<dim3(FF0 / 32, D0 / 32, L0), 256, 0, stream>>>(
      lin1_w, lin1_wt, D0, FF0);
  k_cvt_transpose<<<dim3(D0 / 32, FF0 / 32, L0), 256, 0, stream>>>(
      lin2_w, lin2_wt, FF0, D0);
  k_cvt_transpose<<<dim3((CLS0 + 31) / 32, D0 / 32, 1), 256, 0, stream>>>(
      fc_w, fc_wt, D0, CLS0);
  k_zero_bf16<<<g1((size_t)BH * HD0 * KP, 256), 256, 0, stream>>>(
      vvt, (size_t)BH * HD0 * KP);
  k_zero_bf16<<<g1((size_t)BH * N0 * KP, 256), 256, 0, stream>>>(
      attnbf, (size_t)BH * N0 * KP);

  k_add_pos<<<g1(BND, 256), 256, 0, stream>>>(x, pos, srcA);

  for (int l = 0; l < L0; ++l) {
    k_layernorm<<<ROWS, 256, 0, stream>>>(srcA, nullptr, pre_w + l * D0,
                                          pre_b + l * D0, nullptr, bufY, D0);
    // qkv = y @ qkv_w + b : 6272 x 2304 x 768
    k_gemm<<<dim3(3 * D0 / GM_BN, ROWS / GM_BM), 256, 0, stream>>>(
        bufY, qkv_wt + (size_t)l * 3 * D0 * D0, qkv_b + (size_t)l * 3 * D0,
        nullptr, bufQKV, ROWS, 3 * D0, D0, 0, 0);
    k_qk_prep<<<B0 * H0 * N0, 64, 0, stream>>>(bufQKV, qq, kk, vvt);
    // attn_raw = qq @ kk^T : per (b,h) 196x196x64
    k_gemm_batched<<<dim3(4, 4, BH), 256, 0, stream>>>(
        qq, kk, attn, N0, N0, HD0, (long)N0 * HD0, (long)N0 * HD0,
        (long)N0 * N0, 0);
    k_zero<<<1, 64, 0, stream>>>(stats, 64);
    k_attn_stats<<<dim3(128, B0), 256, 0, stream>>>(attn, stats);
    k_attn_norm<<<g1((size_t)B0 * HNN, 256), 256, 0, stream>>>(
        attn, stats, aln_w + (size_t)l * HNN, aln_b + (size_t)l * HNN, attnbf);
    // out_heads = attn @ v : per (b,h) 196x64x(196 padded 224), bufY reused
    k_gemm_batched<<<dim3(1, 4, BH), 256, 0, stream>>>(
        attnbf, vvt, bufY, N0, HD0, KP, (long)N0 * KP, (long)HD0 * KP,
        (long)N0 * HD0, 1);
    k_merge_heads<<<g1(BND, 256), 256, 0, stream>>>(bufY, qq);
    // proj (fp32 out reuses bufQKV)
    k_gemm<<<dim3(D0 / GM_BN, ROWS / GM_BM), 256, 0, stream>>>(
        qq, proj_wt + (size_t)l * D0 * D0, proj_b + (size_t)l * D0, nullptr,
        bufQKV, ROWS, D0, D0, 0, 0);
    // srcB = LN(srcA + proj), fp32 + bf16
    k_layernorm<<<ROWS, 256, 0, stream>>>(srcA, bufQKV, n1_w + l * D0,
                                          n1_b + l * D0, srcB, srcBb, D0);
    // ff = gelu(srcB @ lin1 + b1) -> bf16
    k_gemm<<<dim3(FF0 / GM_BN, ROWS / GM_BM), 256, 0, stream>>>(
        srcBb, lin1_wt + (size_t)l * FF0 * D0, lin1_b + (size_t)l * FF0,
        nullptr, bufFF, ROWS, FF0, D0, 1, 1);
    // srcA = srcB + ff @ lin2 + b2
    k_gemm<<<dim3(D0 / GM_BN, ROWS / GM_BM), 256, 0, stream>>>(
        bufFF, lin2_wt + (size_t)l * D0 * FF0, lin2_b + (size_t)l * D0, srcB,
        srcA, ROWS, D0, FF0, 0, 0);
  }

  // final LN (fp32 xf reuses bufQKV), pooling, classifier
  k_layernorm<<<ROWS, 256, 0, stream>>>(srcA, nullptr, fin_w, fin_b, bufQKV,
                                        nullptr, D0);
  k_pool_score<<<ROWS, 256, 0, stream>>>(bufQKV, pool_w, pool_b, bufS);
  k_softmax_pool<<<B0, 256, 0, stream>>>(bufQKV, bufS, pooled);
  k_gemm<<<dim3((CLS0 + GM_BN - 1) / GM_BN, 1), 256, 0, stream>>>(
      pooled, fc_wt, fc_b, nullptr, logits, B0, CLS0, D0, 0, 0);
}
